// PointNet2FeaturePropagator_6322191859821
// MI455X (gfx1250) — matrix operand order repaired and checked by reading, not compiled, not run on hardware
//
#include <hip/hip_runtime.h>
#include <hip/hip_bf16.h>
#include <stdint.h>

// ---------------------------------------------------------------------------
// Types for CDNA5 WMMA (wave32): bf16 A/B fragments (16 elems/lane = 8 VGPRs),
// f32 C/D accumulator (8 VGPRs).
// ---------------------------------------------------------------------------
typedef __attribute__((ext_vector_type(16))) __bf16 v16bf;
typedef __attribute__((ext_vector_type(8)))  float  v8f;

union Frag16 {
    v16bf v;
    uint4 q[2];
};

// Problem sizes (fixed by the reference setup_inputs()).
#define BATCH 8
#define NPTS  8192
#define MPREV 2048
#define CF    256
#define CP    512
#define K0DIM 768   // CP + CF
#define CO0   256
#define CO1   128

// GEMM macro-tile config: block = 8 waves; block tile = 32co x 256n;
// wave tile = 32co x 32n = 2x2 WMMA tiles. W panel staged in LDS with a
// padded row stride (K+8 bf16 = +16B) so the 16 lanes of a b128 A-fragment
// read hit 16 disjoint 4-bank groups (no LDS bank conflicts).
#define KMAX     K0DIM
#define LDS_PAD  8

// ---------------------------------------------------------------------------
// Kernel 1: three_nn — brute-force 3-NN with xyz_prev tile in LDS.
// grid: (NPTS/256, BATCH), block: 256
// ---------------------------------------------------------------------------
__global__ __launch_bounds__(256) void k_three_nn(
    const float* __restrict__ xyz,     // (B, N, 3)
    const float* __restrict__ xyzp,    // (B, M, 3)
    int*   __restrict__ idx,           // (B, N, 3)
    float* __restrict__ wgt)           // (B, N, 3)
{
    __shared__ float sp[MPREV * 3];    // 24 KB of the 320 KB WGP LDS
    const int b = blockIdx.y;
    const float* P = xyzp + (size_t)b * MPREV * 3;
    for (int i = threadIdx.x; i < MPREV * 3; i += blockDim.x) sp[i] = P[i];
    __syncthreads();

    const int n = blockIdx.x * blockDim.x + threadIdx.x;
    const float* q = xyz + ((size_t)b * NPTS + n) * 3;
    const float x = q[0], y = q[1], z = q[2];

    float d0 = 3.4e38f, d1 = 3.4e38f, d2 = 3.4e38f;
    int   i0 = 0, i1 = 0, i2 = 0;
    for (int j = 0; j < MPREV; ++j) {
        const float dx = x - sp[3 * j + 0];
        const float dy = y - sp[3 * j + 1];
        const float dz = z - sp[3 * j + 2];
        const float d  = dx * dx + dy * dy + dz * dz;
        if (d < d2) {
            if (d < d1) {
                d2 = d1; i2 = i1;
                if (d < d0) { d1 = d0; i1 = i0; d0 = d; i0 = j; }
                else        { d1 = d;  i1 = j; }
            } else { d2 = d; i2 = j; }
        }
    }
    const float a0 = 1.0f / (d0 + 1e-8f);
    const float a1 = 1.0f / (d1 + 1e-8f);
    const float a2 = 1.0f / (d2 + 1e-8f);
    const float inv = 1.0f / (a0 + a1 + a2);
    const size_t o = ((size_t)b * NPTS + n) * 3;
    idx[o + 0] = i0; idx[o + 1] = i1; idx[o + 2] = i2;
    wgt[o + 0] = a0 * inv; wgt[o + 1] = a1 * inv; wgt[o + 2] = a2 * inv;
}

// ---------------------------------------------------------------------------
// Kernel 2: three_interpolate + concat -> X0 (B, N, 768) bf16.
// features_prev (33 MB) is L2-resident, so the strided gathers are cheap.
// grid: (NPTS, BATCH), block: 256
// ---------------------------------------------------------------------------
__global__ __launch_bounds__(256) void k_interp_concat(
    const float* __restrict__ fprev,   // (B, Cp, M)
    const float* __restrict__ feat,    // (B, C,  N)
    const int*   __restrict__ idx,     // (B, N, 3)
    const float* __restrict__ wgt,     // (B, N, 3)
    __bf16*      __restrict__ X0)      // (B, N, Cp+C)
{
    const int b = blockIdx.y;
    const int n = blockIdx.x;
    const size_t bn = (size_t)b * NPTS + n;
    const int*   id = idx + bn * 3;
    const float* w  = wgt + bn * 3;
    const int   j0 = id[0], j1 = id[1], j2 = id[2];
    const float w0 = w[0],  w1 = w[1],  w2 = w[2];

    __bf16* xo = X0 + bn * K0DIM;
    const float* fp = fprev + (size_t)b * CP * MPREV;
    for (int k = threadIdx.x; k < CP; k += blockDim.x) {
        const float* row = fp + (size_t)k * MPREV;
        xo[k] = (__bf16)(w0 * row[j0] + w1 * row[j1] + w2 * row[j2]);
    }
    const float* f = feat + (size_t)b * CF * NPTS;
    for (int k = threadIdx.x; k < CF; k += blockDim.x)
        xo[CP + k] = (__bf16)f[(size_t)k * NPTS + n];
}

// ---------------------------------------------------------------------------
// Weight fp32 -> bf16 conversion (one-shot).
// ---------------------------------------------------------------------------
__global__ __launch_bounds__(256) void k_cvt_w(
    const float* __restrict__ W0, const float* __restrict__ W1,
    __bf16* __restrict__ W0b, __bf16* __restrict__ W1b, int n0, int n1)
{
    const int i = blockIdx.x * blockDim.x + threadIdx.x;
    if (i < n0)           W0b[i]      = (__bf16)W0[i];
    else if (i < n0 + n1) W1b[i - n0] = (__bf16)W1[i - n0];
}

__global__ __launch_bounds__(256) void k_zero(float* __restrict__ p, int n)
{
    const int i = blockIdx.x * blockDim.x + threadIdx.x;
    if (i < n) p[i] = 0.0f;
}

// ---------------------------------------------------------------------------
// Kernel 3/5: bf16 GEMM with LDS-staged weight panel (async copy) and 2x2
// register blocking per wave.
//   Y[b, n, co] = sum_k W[co, k] * X[b, n, k]
// Block (8 waves) computes a 32co x 256n macro-tile; the 32 x K weight panel
// is copied Global->LDS with global_load_async_to_lds_b128 (ASYNCcnt), then
// each wave runs 2co x 2n WMMA tiles: 4 ds_load_b128 + 4 global_load_b128
// per 4 v_wmma (K step 32). BN sum/sumsq accumulated via width-16 shfl_xor
// reduction + one atomicAdd per channel per wave.
// grid: B*(CO/32)*(N/256) blocks, block: 256 (8 waves)
// ---------------------------------------------------------------------------
__global__ __launch_bounds__(256) void k_gemm_bf16(
    const __bf16* __restrict__ W,      // (CO, K)
    const __bf16* __restrict__ X,      // (B, N, K)
    float* __restrict__ Y,             // (B, N, CO)
    float* __restrict__ sum,           // (CO)
    float* __restrict__ sumsq,         // (CO)
    int CO, int K, int N)
{
    // Single shared object -> LDS offset 0; async-copy target addresses are
    // plain byte offsets into this panel.
    __shared__ __attribute__((aligned(16))) __bf16 sW[32 * (KMAX + LDS_PAD)];

    // CRITICAL: make sW's address escape into opaque asm. The async-copy
    // instructions below address LDS only via raw integer offsets, so without
    // this the compiler proves sW is never stored to, folds the ds_load reads
    // of the panel to undef, and hoists the A fragments out of the K loop
    // (observed in round-2 disassembly: loop-invariant v[0:7] A operand).
    asm volatile("" :: "v"((const void*)sW) : "memory");

    const int w    = threadIdx.x >> 5;       // wave id 0..7
    const int lane = threadIdx.x & 31;
    const int half = lane >> 4;              // lane group 0-15 / 16-31
    const int ml   = lane & 15;

    const int nBlocks  = N / 256;
    const int coBlocks = CO / 32;
    const int perB     = coBlocks * nBlocks;
    const int b   = blockIdx.x / perB;
    const int rem = blockIdx.x % perB;
    const int co0 = (rem / nBlocks) * 32;
    const int n0  = (rem % nBlocks) * 256 + w * 32;

    const int strideA = K + LDS_PAD;

    // ---- Stage the 32 x K weight panel into LDS via async copy -------------
    {
        const int chunksPerRow = K >> 3;           // 16B chunks per row
        const int totalChunks  = 32 * chunksPerRow;
        for (int c = threadIdx.x; c < totalChunks; c += 256) {
            const int r  = c / chunksPerRow;
            const int kc = (c % chunksPerRow) << 3;
            const unsigned ldsAddr = (unsigned)((r * strideA + kc) * 2);
            const __bf16* src = W + (size_t)(co0 + r) * K + kc;
            asm volatile("global_load_async_to_lds_b128 %0, %1, off"
                         :: "v"(ldsAddr), "v"(src) : "memory");
        }
        asm volatile("s_wait_asynccnt 0x0" ::: "memory");
        __syncthreads();
    }

    // ---- Main K loop: 2co x 2n WMMA tiles per wave -------------------------
    const __bf16* A0  = &sW[(size_t)ml * strideA];          // co rows  0..15
    const __bf16* A1  = &sW[(size_t)(16 + ml) * strideA];   // co rows 16..31
    const __bf16* Xr0 = X + ((size_t)b * N + n0 + ml) * K;       // n  0..15
    const __bf16* Xr1 = X + ((size_t)b * N + n0 + 16 + ml) * K;  // n 16..31

    v8f acc[2][2];
    acc[0][0] = (v8f){}; acc[0][1] = (v8f){};
    acc[1][0] = (v8f){}; acc[1][1] = (v8f){};

    for (int k0 = 0; k0 < K; k0 += 32) {
        Frag16 a0, a1, bx0, bx1;
        // 16-bit A layout: lanes 0-15 hold K=[k0,k0+8)+[k0+16,k0+24);
        //                  lanes 16-31 hold K=[k0+8,k0+16)+[k0+24,k0+32).
        a0.q[0]  = *(const uint4*)(A0 + k0 + half * 8);
        a0.q[1]  = *(const uint4*)(A0 + k0 + 16 + half * 8);
        a1.q[0]  = *(const uint4*)(A1 + k0 + half * 8);
        a1.q[1]  = *(const uint4*)(A1 + k0 + 16 + half * 8);
        // 16-bit B layout: same K order as A (lanes 0-15 K=[k0,k0+8)+[k0+16,k0+24); lanes 16-31 K=[k0+8,k0+16)+[k0+24,k0+32)).
        bx0.q[0] = *(const uint4*)(Xr0 + k0 + half * 8);
        bx0.q[1] = *(const uint4*)(Xr0 + k0 + 16 + half * 8);
        bx1.q[0] = *(const uint4*)(Xr1 + k0 + half * 8);
        bx1.q[1] = *(const uint4*)(Xr1 + k0 + 16 + half * 8);

        acc[0][0] = __builtin_amdgcn_wmma_f32_16x16x32_bf16(
            false, a0.v, false, bx0.v, (short)0, acc[0][0], false, false);
        acc[0][1] = __builtin_amdgcn_wmma_f32_16x16x32_bf16(
            false, a0.v, false, bx1.v, (short)0, acc[0][1], false, false);
        acc[1][0] = __builtin_amdgcn_wmma_f32_16x16x32_bf16(
            false, a1.v, false, bx0.v, (short)0, acc[1][0], false, false);
        acc[1][1] = __builtin_amdgcn_wmma_f32_16x16x32_bf16(
            false, a1.v, false, bx1.v, (short)0, acc[1][1], false, false);
    }

    // ---- Store: C/D layout VGPR r -> co = base + half*8 + r, lane%16 -> n --
#pragma unroll
    for (int ct = 0; ct < 2; ++ct) {
#pragma unroll
        for (int nt = 0; nt < 2; ++nt) {
            float* yp = Y + ((size_t)b * N + n0 + nt * 16 + ml) * CO
                          + co0 + ct * 16 + half * 8;
            const v8f& a = acc[ct][nt];
            ((float4*)yp)[0] = make_float4(a[0], a[1], a[2], a[3]);
            ((float4*)yp)[1] = make_float4(a[4], a[5], a[6], a[7]);
        }
    }

    // ---- BN partial stats: fold the two n-tiles, reduce across 16 lanes ----
#pragma unroll
    for (int ct = 0; ct < 2; ++ct) {
#pragma unroll
        for (int r = 0; r < 8; ++r) {
            float s  = acc[ct][0][r] + acc[ct][1][r];
            float q2 = acc[ct][0][r] * acc[ct][0][r]
                     + acc[ct][1][r] * acc[ct][1][r];
            for (int m = 1; m < 16; m <<= 1) {
                s  += __shfl_xor(s,  m, 16);
                q2 += __shfl_xor(q2, m, 16);
            }
            if (ml == 0) {
                const int co = co0 + ct * 16 + half * 8 + r;
                atomicAdd(&sum[co],   s);
                atomicAdd(&sumsq[co], q2);
            }
        }
    }
}

// ---------------------------------------------------------------------------
// Kernel 4: BN (training stats) + ReLU -> bf16 activations for the next GEMM.
// ---------------------------------------------------------------------------
__global__ __launch_bounds__(256) void k_bn_relu_bf16(
    const float* __restrict__ Y,       // (B, N, CO)
    const float* __restrict__ g, const float* __restrict__ be,
    const float* __restrict__ sum, const float* __restrict__ sumsq,
    __bf16* __restrict__ H, int CO, float invCnt, size_t total)
{
    const size_t i = (size_t)blockIdx.x * blockDim.x + threadIdx.x;
    if (i >= total) return;
    const int co = (int)(i % CO);
    const float mean = sum[co] * invCnt;
    const float var  = sumsq[co] * invCnt - mean * mean;
    float v = (Y[i] - mean) * rsqrtf(var + 1e-5f);
    v = v * g[co] + be[co];
    H[i] = (__bf16)fmaxf(v, 0.0f);
}

// ---------------------------------------------------------------------------
// Kernel 6: BN + ReLU + transpose to the reference output layout (B, CO, N) f32.
// ---------------------------------------------------------------------------
__global__ __launch_bounds__(256) void k_bn_relu_out(
    const float* __restrict__ Y,       // (B, N, CO)
    const float* __restrict__ g, const float* __restrict__ be,
    const float* __restrict__ sum, const float* __restrict__ sumsq,
    float* __restrict__ out,           // (B, CO, N)
    int CO, float invCnt, size_t total)
{
    const size_t i = (size_t)blockIdx.x * blockDim.x + threadIdx.x;
    if (i >= total) return;
    const int co  = (int)(i % CO);
    const size_t bn = i / CO;
    const int n = (int)(bn % NPTS);
    const int b = (int)(bn / NPTS);
    const float mean = sum[co] * invCnt;
    const float var  = sumsq[co] * invCnt - mean * mean;
    float v = (Y[i] - mean) * rsqrtf(var + 1e-5f);
    v = v * g[co] + be[co];
    out[((size_t)b * CO + co) * NPTS + n] = fmaxf(v, 0.0f);
}

// ---------------------------------------------------------------------------
// Host launcher — workspace layout (bytes, all 256-aligned):
// ---------------------------------------------------------------------------
#define OFF_IDX   ((size_t)0)                       // int  [B*N*3]     786432 B
#define OFF_WGT   (OFF_IDX  + 786432)               // f32  [B*N*3]     786432 B
#define OFF_X0    (OFF_WGT  + 786432)               // bf16 [B*N*768]   100663296 B
#define OFF_W0B   (OFF_X0   + 100663296)            // bf16 [256*768]   393216 B
#define OFF_W1B   (OFF_W0B  + 393216)               // bf16 [128*256]   65536 B
#define OFF_Y0    (OFF_W1B  + 65536)                // f32  [B*N*256]   67108864 B
#define OFF_H     (OFF_Y0   + 67108864)             // bf16 [B*N*256]   33554432 B
#define OFF_Y1    (OFF_H    + 33554432)             // f32  [B*N*128]   33554432 B
#define OFF_STATS (OFF_Y1   + 33554432)             // f32  [768]

extern "C" void kernel_launch(void* const* d_in, const int* in_sizes, int n_in,
                              void* d_out, int out_size, void* d_ws, size_t ws_size,
                              hipStream_t stream) {
    const float* xyz   = (const float*)d_in[0];
    const float* xyzp  = (const float*)d_in[1];
    const float* feat  = (const float*)d_in[2];
    const float* fprev = (const float*)d_in[3];
    const float* W0 = (const float*)d_in[4];
    // d_in[5] = b0 (cancelled exactly by BN mean subtraction — skipped)
    const float* g0  = (const float*)d_in[6];
    const float* be0 = (const float*)d_in[7];
    const float* W1 = (const float*)d_in[8];
    // d_in[9] = b1 (skipped, same reason)
    const float* g1  = (const float*)d_in[10];
    const float* be1 = (const float*)d_in[11];
    float* out = (float*)d_out;

    char* ws = (char*)d_ws;
    int*    idx  = (int*)  (ws + OFF_IDX);
    float*  wgt  = (float*)(ws + OFF_WGT);
    __bf16* X0   = (__bf16*)(ws + OFF_X0);
    __bf16* W0b  = (__bf16*)(ws + OFF_W0B);
    __bf16* W1b  = (__bf16*)(ws + OFF_W1B);
    float*  Y0   = (float*)(ws + OFF_Y0);
    __bf16* H    = (__bf16*)(ws + OFF_H);
    float*  Y1   = (float*)(ws + OFF_Y1);
    float*  stats = (float*)(ws + OFF_STATS);
    float* sum0 = stats;        // 256
    float* sq0  = stats + 256;  // 256
    float* sum1 = stats + 512;  // 128
    float* sq1  = stats + 640;  // 128

    const float invCnt = 1.0f / (float)(BATCH * NPTS);

    // Stats must be zero every call (deterministic, graph-replay safe).
    k_zero<<<3, 256, 0, stream>>>(stats, 768);

    // Weights -> bf16.
    k_cvt_w<<<(CO0 * K0DIM + CO1 * CO0 + 255) / 256, 256, 0, stream>>>(
        W0, W1, W0b, W1b, CO0 * K0DIM, CO1 * CO0);

    // 3-NN + interpolation weights.
    k_three_nn<<<dim3(NPTS / 256, BATCH), 256, 0, stream>>>(xyz, xyzp, idx, wgt);

    // Interpolate + concat -> X0 (B, N, 768) bf16.
    k_interp_concat<<<dim3(NPTS, BATCH), 256, 0, stream>>>(fprev, feat, idx, wgt, X0);

    // GEMM0: (256 x 768) x (768 x B*N) — B*(CO/32)*(N/256) = 2048 blocks.
    k_gemm_bf16<<<BATCH * (CO0 / 32) * (NPTS / 256), 256, 0, stream>>>(
        W0b, X0, Y0, sum0, sq0, CO0, K0DIM, NPTS);

    // BN0 + ReLU -> bf16 H.
    {
        const size_t total = (size_t)BATCH * NPTS * CO0;
        k_bn_relu_bf16<<<(unsigned)((total + 255) / 256), 256, 0, stream>>>(
            Y0, g0, be0, sum0, sq0, H, CO0, invCnt, total);
    }

    // GEMM1: (128 x 256) x (256 x B*N) — 1024 blocks.
    k_gemm_bf16<<<BATCH * (CO1 / 32) * (NPTS / 256), 256, 0, stream>>>(
        W1b, H, Y1, sum1, sq1, CO1, CO0, NPTS);

    // BN1 + ReLU + transpose to (B, 128, N) f32 output.
    {
        const size_t total = (size_t)BATCH * NPTS * CO1;
        k_bn_relu_out<<<(unsigned)((total + 255) / 256), 256, 0, stream>>>(
            Y1, g1, be1, sum1, sq1, out, CO1, invCnt, total);
    }
    (void)in_sizes; (void)n_in; (void)out_size; (void)ws_size;
}
